// ProtoInstanceMultiCrossAttention_90426241450028
// MI455X (gfx1250) — matrix operand order repaired and checked by reading, not compiled, not run on hardware
//
#include <hip/hip_runtime.h>
#include <hip/hip_bf16.h>

// ---------------------------------------------------------------------------
// Types for WMMA (CDNA5 / gfx1250, wave32)
// ---------------------------------------------------------------------------
typedef __attribute__((ext_vector_type(16))) __bf16 v16bf;
typedef __attribute__((ext_vector_type(8)))  float  v8f;

union FragBF {            // 16 bf16 elements = one WMMA A/B operand (8 VGPRs)
    v16bf v;
    uint4 q[2];
};
union U8h { uint4 q; unsigned short h[8]; };
union U4h { uint2 q; unsigned short h[4]; };

__device__ __forceinline__ unsigned short f32_to_bf16(float f) {
    unsigned u = __float_as_uint(f);
    unsigned r = u + 0x7FFFu + ((u >> 16) & 1u);   // round-to-nearest-even
    return (unsigned short)(r >> 16);
}
__device__ __forceinline__ float bf16_to_f32(unsigned short b) {
    return __uint_as_float(((unsigned)b) << 16);
}

#define WMMA_BF16(A, B, C) \
    __builtin_amdgcn_wmma_f32_16x16x32_bf16(false, (A), false, (B), (short)0, (C), false, false)

// Problem constants (C=10, S=10, NQ=30, H=1024)
#define HDIM 1024
#define NSUP 100          // C*S support rows
#define NSUP_PAD 112      // padded to 7 M-tiles of 16
#define NQRY 300          // C*NQ query rows
#define NQRY_PAD 304
#define NROWS 400

// ---------------------------------------------------------------------------
// Gather x -> (support rows 0..99, query rows 100..399), convert f32 -> bf16
// ---------------------------------------------------------------------------
__global__ void gather_x_bf16_kernel(const float* __restrict__ x,
                                     unsigned short* __restrict__ Xb) {
    int r = blockIdx.x;                       // 0..399 dest row
    int src;
    if (r < NSUP) src = (r / 10) * 40 + (r % 10);                // support
    else { int rq = r - NSUP; src = (rq / 30) * 40 + 10 + (rq % 30); } // query
    const float* in = x + (size_t)src * HDIM;
    unsigned short* out = Xb + (size_t)r * HDIM;
    for (int i = threadIdx.x * 4; i < HDIM; i += blockDim.x * 4) {
        float4 v = *(const float4*)(in + i);
        U4h o;
        o.h[0] = f32_to_bf16(v.x); o.h[1] = f32_to_bf16(v.y);
        o.h[2] = f32_to_bf16(v.z); o.h[3] = f32_to_bf16(v.w);
        *(uint2*)(out + i) = o.q;
    }
}

// ---------------------------------------------------------------------------
// Transpose 1024x1024 f32 -> bf16 (out[d][h] = in[h][d]); B operand wants N x K
// ---------------------------------------------------------------------------
__global__ void transpose_bf16_kernel(const float* __restrict__ in,
                                      unsigned short* __restrict__ out) {
    __shared__ float tile[32][33];
    int tx = threadIdx.x, ty = threadIdx.y;       // (32, 8)
    int bx = blockIdx.x * 32, by = blockIdx.y * 32;
#pragma unroll
    for (int k = 0; k < 4; ++k)
        tile[ty + k * 8][tx] = in[(size_t)(by + ty + k * 8) * HDIM + bx + tx];
    __syncthreads();
#pragma unroll
    for (int k = 0; k < 4; ++k)
        out[(size_t)(bx + ty + k * 8) * HDIM + by + tx] =
            f32_to_bf16(tile[tx][ty + k * 8]);
}

// ---------------------------------------------------------------------------
// Generic bf16 WMMA GEMM: C[M x 1024] = A[M x 1024] @ BT^T (+bias)
//   A   : M x 1024 bf16 row-major (row reads clamped to M-1 for padding)
//   BT  : 1024 x 1024 bf16 row-major, row n = column n of B (K contiguous)
//   Cf  : optional f32 out, Cb : optional bf16 out (padded rows all written)
// block = 128 threads (4 waves); wave w covers 4 n-tiles (64 cols).
// grid  = (Mpad/16, 1024/256)
// ---------------------------------------------------------------------------
__global__ __launch_bounds__(128)
void gemm_bf16_kernel(const unsigned short* __restrict__ A,
                      const unsigned short* __restrict__ BT,
                      const float* __restrict__ bias,
                      float* __restrict__ Cf,
                      unsigned short* __restrict__ Cb,
                      int M) {
    const int lane = threadIdx.x & 31;
    const int wave = threadIdx.x >> 5;
    const int l15  = lane & 15;
    const int h    = lane >> 4;
    const int mt   = blockIdx.x;
    const int ncol0 = blockIdx.y * 256 + wave * 64;

    int arow = mt * 16 + l15;
    if (arow >= M) arow = M - 1;
    const unsigned short* aptr = A + (size_t)arow * HDIM;

    v8f acc[4] = {};
    for (int k0 = 0; k0 < HDIM; k0 += 32) {
        FragBF fa;
        fa.q[0] = *(const uint4*)(aptr + k0 + 16 * h);       // K 16h..16h+7
        fa.q[1] = *(const uint4*)(aptr + k0 + 16 * h + 8);   // K 16h+8..+15
#pragma unroll
        for (int nt = 0; nt < 4; ++nt) {
            const unsigned short* bptr =
                BT + (size_t)(ncol0 + nt * 16 + l15) * HDIM + k0 + 16 * h;
            FragBF fb;
            fb.q[0] = *(const uint4*)(bptr);                  // K 16h..16h+7
            fb.q[1] = *(const uint4*)(bptr + 8);              // K 16h+8..+15
            acc[nt] = WMMA_BF16(fa.v, fb.v, acc[nt]);
        }
    }
#pragma unroll
    for (int nt = 0; nt < 4; ++nt) {
        int col = ncol0 + nt * 16 + l15;
        float bv = bias ? bias[col] : 0.0f;
#pragma unroll
        for (int j = 0; j < 8; ++j) {
            int row = mt * 16 + j + 8 * h;
            float v = acc[nt][j] + bv;
            if (Cf) Cf[(size_t)row * HDIM + col] = v;
            if (Cb) Cb[(size_t)row * HDIM + col] = f32_to_bf16(v);
        }
    }
}

// ---------------------------------------------------------------------------
// Phase 2: for each q, per 512-col block:
//   T = D1@Wc + D2@Wd  (D1=|sf-qf[q]|, D2=sf*qf[q], built per 64-K chunk in LDS)
//   tsum_part[q][nb][row] = sum_{cols in block} tanh(T + a_s + a_q[q] + b2)
// grid (300, 2), block 512 (16 waves, two 16-col tiles per wave -> A reuse x2)
// ---------------------------------------------------------------------------
__global__ __launch_bounds__(512)
void phase2_kernel(const unsigned short* __restrict__ Fb,   // 400 x 1024 bf16
                   const unsigned short* __restrict__ WcT,  // 1024 x 1024 bf16
                   const unsigned short* __restrict__ WdT,
                   const float* __restrict__ as_buf,        // 112 x 1024 f32
                   const float* __restrict__ aq_buf,        // 304 x 1024 f32
                   const float* __restrict__ b2,
                   float* __restrict__ tsum_part) {         // 300 x 2 x 112
    const int q    = blockIdx.x;
    const int nb   = blockIdx.y;          // 0..1
    const int tid  = threadIdx.x;         // 0..511
    const int lane = tid & 31;
    const int wave = tid >> 5;            // 0..15
    const int l15  = lane & 15;
    const int h    = lane >> 4;

    __shared__ __align__(16) unsigned short d1s[NSUP_PAD * 64];
    __shared__ __align__(16) unsigned short d2s[NSUP_PAD * 64];
    __shared__ float qch[64];
    __shared__ float tsum_l[NSUP_PAD];

    if (tid < NSUP_PAD) tsum_l[tid] = 0.0f;

    const unsigned short* qrow = Fb + (size_t)(NSUP + q) * HDIM;
    const int ncol = nb * 512 + wave * 32;            // wave's 2 n-tiles

    v8f acc[7][2] = {};
    for (int k0 = 0; k0 < HDIM; k0 += 64) {
        __syncthreads();                   // prior chunk consumed / init done
        if (tid < 64) qch[tid] = bf16_to_f32(qrow[k0 + tid]);
        __syncthreads();
        // build D1/D2 chunk: 112 rows x 64 K, 16 elements per thread
        if (tid < NSUP_PAD * 4) {
            int r  = tid >> 2;
            int kk = (tid & 3) * 16;
            U8h o1a, o1b, o2a, o2b;
            if (r < NSUP) {
                U8h sa, sb;
                const unsigned short* sp = Fb + (size_t)r * HDIM + k0 + kk;
                sa.q = *(const uint4*)(sp);
                sb.q = *(const uint4*)(sp + 8);
#pragma unroll
                for (int e = 0; e < 8; ++e) {
                    float sv = bf16_to_f32(sa.h[e]);
                    float qv = qch[kk + e];
                    o1a.h[e] = f32_to_bf16(fabsf(sv - qv));
                    o2a.h[e] = f32_to_bf16(sv * qv);
                    float sv2 = bf16_to_f32(sb.h[e]);
                    float qv2 = qch[kk + 8 + e];
                    o1b.h[e] = f32_to_bf16(fabsf(sv2 - qv2));
                    o2b.h[e] = f32_to_bf16(sv2 * qv2);
                }
            } else {
                o1a.q = make_uint4(0, 0, 0, 0); o1b.q = o1a.q;
                o2a.q = o1a.q;                  o2b.q = o1a.q;
            }
            *(uint4*)(d1s + r * 64 + kk)     = o1a.q;
            *(uint4*)(d1s + r * 64 + kk + 8) = o1b.q;
            *(uint4*)(d2s + r * 64 + kk)     = o2a.q;
            *(uint4*)(d2s + r * 64 + kk + 8) = o2b.q;
        }
        __syncthreads();
#pragma unroll
        for (int ks = 0; ks < 2; ++ks) {              // two 32-K steps
            const int kb = k0 + ks * 32;
            FragBF fc[2], fd[2];
#pragma unroll
            for (int nt = 0; nt < 2; ++nt) {
                const unsigned short* cp =
                    WcT + (size_t)(ncol + nt * 16 + l15) * HDIM + kb + 16 * h;
                fc[nt].q[0] = *(const uint4*)(cp);
                fc[nt].q[1] = *(const uint4*)(cp + 8);
                const unsigned short* dp =
                    WdT + (size_t)(ncol + nt * 16 + l15) * HDIM + kb + 16 * h;
                fd[nt].q[0] = *(const uint4*)(dp);
                fd[nt].q[1] = *(const uint4*)(dp + 8);
            }
#pragma unroll
            for (int m = 0; m < 7; ++m) {
                FragBF fa1, fa2;
                const unsigned short* l1 = d1s + (m * 16 + l15) * 64 + ks * 32;
                fa1.q[0] = *(const uint4*)(l1 + 16 * h);
                fa1.q[1] = *(const uint4*)(l1 + 16 * h + 8);
                acc[m][0] = WMMA_BF16(fa1.v, fc[0].v, acc[m][0]);
                acc[m][1] = WMMA_BF16(fa1.v, fc[1].v, acc[m][1]);
                const unsigned short* l2 = d2s + (m * 16 + l15) * 64 + ks * 32;
                fa2.q[0] = *(const uint4*)(l2 + 16 * h);
                fa2.q[1] = *(const uint4*)(l2 + 16 * h + 8);
                acc[m][0] = WMMA_BF16(fa2.v, fd[0].v, acc[m][0]);
                acc[m][1] = WMMA_BF16(fa2.v, fd[1].v, acc[m][1]);
            }
        }
    }
    // epilogue: + a_s + a_q + b2, tanh, reduce over each tile's 16 columns
#pragma unroll
    for (int nt = 0; nt < 2; ++nt) {
        const int col = ncol + nt * 16 + l15;
        const float qb = aq_buf[(size_t)q * HDIM + col] + b2[col];
#pragma unroll
        for (int m = 0; m < 7; ++m) {
#pragma unroll
            for (int j = 0; j < 8; ++j) {
                int row = m * 16 + j + 8 * h;
                float v = acc[m][nt][j] + as_buf[(size_t)row * HDIM + col] + qb;
                float t = tanhf(v);
                t += __shfl_xor(t, 1, 32);
                t += __shfl_xor(t, 2, 32);
                t += __shfl_xor(t, 4, 32);
                t += __shfl_xor(t, 8, 32);      // sum over col bits (l&15)
                if (l15 == 0) atomicAdd(&tsum_l[row], t);
            }
        }
    }
    __syncthreads();
    if (tid < NSUP_PAD)
        tsum_part[((size_t)q * 2 + nb) * NSUP_PAD + tid] = tsum_l[tid];
}

// ---------------------------------------------------------------------------
// Phase 3: softmax over S, proto = score-weighted support, squared distances
// grid 300 (one q per block), block 128
// ---------------------------------------------------------------------------
__global__ __launch_bounds__(128)
void phase3_kernel(const float* __restrict__ x,          // original 400 x 1024
                   const float* __restrict__ tsum_part,  // 300 x 2 x 112
                   float* __restrict__ dists) {          // 300 x 10
    const int q = blockIdx.x;
    const int tid = threadIdx.x;
    __shared__ float ts[NSUP];
    __shared__ float sc[10][10];
    __shared__ float red[128];

    if (tid < NSUP) {
        float v = 0.0f;
#pragma unroll
        for (int nb = 0; nb < 2; ++nb)
            v += tsum_part[((size_t)q * 2 + nb) * NSUP_PAD + tid];
        ts[tid] = v;
    }
    __syncthreads();
    if (tid < 10) {                           // softmax over S per class
        float mx = -1e30f;
        for (int s = 0; s < 10; ++s) mx = fmaxf(mx, ts[tid * 10 + s]);
        float e[10], sum = 0.0f;
        for (int s = 0; s < 10; ++s) { e[s] = __expf(ts[tid * 10 + s] - mx); sum += e[s]; }
        float inv = 1.0f / sum;
        for (int s = 0; s < 10; ++s) sc[tid][s] = e[s] * inv;
    }
    __syncthreads();
    const int cq = q / 30, wq = q % 30;
    const float* qrow = x + (size_t)(cq * 40 + 10 + wq) * HDIM;
    for (int c = 0; c < 10; ++c) {
        float part = 0.0f;
        for (int hh = tid; hh < HDIM; hh += 128) {
            float p = 0.0f;
#pragma unroll
            for (int s = 0; s < 10; ++s)
                p += sc[c][s] * x[(size_t)(c * 40 + s) * HDIM + hh];
            float d = p - qrow[hh];
            part += d * d;
        }
        red[tid] = part;
        __syncthreads();
        for (int o = 64; o > 0; o >>= 1) {
            if (tid < o) red[tid] += red[tid + o];
            __syncthreads();
        }
        if (tid == 0) dists[q * 10 + c] = red[0];
        __syncthreads();
    }
}

// ---------------------------------------------------------------------------
// Launch
// ---------------------------------------------------------------------------
extern "C" void kernel_launch(void* const* d_in, const int* in_sizes, int n_in,
                              void* d_out, int out_size, void* d_ws, size_t ws_size,
                              hipStream_t stream) {
    const float* x  = (const float*)d_in[0];
    const float* W1 = (const float*)d_in[1];
    const float* b1 = (const float*)d_in[2];
    const float* W2 = (const float*)d_in[3];
    const float* b2 = (const float*)d_in[4];
    float* dout = (float*)d_out;

    char* ws = (char*)d_ws;
    size_t off = 0;
    auto wsalloc = [&](size_t bytes) -> void* {
        void* p = ws + off;
        off = (off + bytes + 255) & ~(size_t)255;
        return p;
    };
    unsigned short* Xb   = (unsigned short*)wsalloc((size_t)NROWS * HDIM * 2);
    unsigned short* W1T  = (unsigned short*)wsalloc((size_t)HDIM * HDIM * 2);
    unsigned short* WsT  = (unsigned short*)wsalloc((size_t)HDIM * HDIM * 2);
    unsigned short* WqT  = (unsigned short*)wsalloc((size_t)HDIM * HDIM * 2);
    unsigned short* WcT  = (unsigned short*)wsalloc((size_t)HDIM * HDIM * 2);
    unsigned short* WdT  = (unsigned short*)wsalloc((size_t)HDIM * HDIM * 2);
    unsigned short* Fb   = (unsigned short*)wsalloc((size_t)NROWS * HDIM * 2);
    float* as_buf        = (float*)wsalloc((size_t)NSUP_PAD * HDIM * 4);
    float* aq_buf        = (float*)wsalloc((size_t)NQRY_PAD * HDIM * 4);
    float* tsum_part     = (float*)wsalloc((size_t)NQRY * 2 * NSUP_PAD * 4);

    // 1. gather + convert x
    gather_x_bf16_kernel<<<NROWS, 256, 0, stream>>>(x, Xb);

    // 2. transpose weights to bf16 N x K layout
    dim3 tb(32, 8), tg(32, 32);
    transpose_bf16_kernel<<<tg, tb, 0, stream>>>(W1, W1T);
    transpose_bf16_kernel<<<tg, tb, 0, stream>>>(W2 + 0 * (size_t)HDIM * HDIM, WsT);
    transpose_bf16_kernel<<<tg, tb, 0, stream>>>(W2 + 1 * (size_t)HDIM * HDIM, WqT);
    transpose_bf16_kernel<<<tg, tb, 0, stream>>>(W2 + 2 * (size_t)HDIM * HDIM, WcT);
    transpose_bf16_kernel<<<tg, tb, 0, stream>>>(W2 + 3 * (size_t)HDIM * HDIM, WdT);

    // 3. F = X @ W1 + b1  (bf16 out, rows: 100 support then 300 query)
    gemm_bf16_kernel<<<dim3(NROWS / 16, 4), 128, 0, stream>>>(
        Xb, W1T, b1, nullptr, Fb, NROWS);
    // 4. a_s = F_s @ Ws ; a_q = F_q @ Wq  (f32 out)
    gemm_bf16_kernel<<<dim3(NSUP_PAD / 16, 4), 128, 0, stream>>>(
        Fb, WsT, nullptr, as_buf, nullptr, NSUP);
    gemm_bf16_kernel<<<dim3(NQRY_PAD / 16, 4), 128, 0, stream>>>(
        Fb + (size_t)NSUP * HDIM, WqT, nullptr, aq_buf, nullptr, NQRY);

    // 5. heavy phase: tanh(D1@Wc + D2@Wd + a_s + a_q + b2) row sums
    phase2_kernel<<<dim3(NQRY, 2), 512, 0, stream>>>(
        Fb, WcT, WdT, as_buf, aq_buf, b2, tsum_part);

    // 6. softmax / proto / distances
    phase3_kernel<<<NQRY, 128, 0, stream>>>(x, tsum_part, dout);
}
